// LinearQuantLoRA_77661598646289
// MI455X (gfx1250) — compile-verified, matrix operand order checked by reading, not run on hardware
//
#include <hip/hip_runtime.h>

// ---------------------------------------------------------------------------
// LinearQuantLoRA on MI455X (gfx1250, wave32, WMMA + async-DMA-to-LDS).
// out[M,N] = fakequant(x/scales) @ quant_w^T + ((..)@right^T)@left^T + bias
// M=32768 (B*S), IN=N=1152, R=64.  A stored as exact-integer f16 (q-zp);
// scale folded into f16 weights; low-rank fused by extending K to 1216.
// Main GEMM: double-buffered global_load_async_to_lds_b128 staging, WMMA
// fed from LDS.
// ---------------------------------------------------------------------------

typedef __attribute__((ext_vector_type(16))) _Float16 v16h;
typedef __attribute__((ext_vector_type(8)))  float    v8f;

constexpr int   Mtot = 32768;       // B*S
constexpr int   IN   = 1152;
constexpr int   OUT  = 1152;
constexpr int   RNK  = 64;
constexpr int   KEXT = IN + RNK;    // 1216, multiple of 32
constexpr float QMAXF = 255.0f;

constexpr int   ROWH  = 40;         // LDS row stride in halfs (80 B, 16B-mult)
constexpr int   TILEH = 128 * ROWH; // halfs per LDS tile buffer (10240 B)

// ---------------- float atomic min/max via sign-split integer atomics ------
__device__ inline void atomicMaxF(float* a, float v) {
  if (v >= 0.0f) atomicMax((int*)a, __float_as_int(v));
  else           atomicMin((unsigned int*)a, __float_as_uint(v));
}
__device__ inline void atomicMinF(float* a, float v) {
  if (v >= 0.0f) atomicMin((int*)a, __float_as_int(v));
  else           atomicMax((unsigned int*)a, __float_as_uint(v));
}

// ---------------- scalar init: s[0]=min, s[1]=max --------------------------
__global__ void k_init(float* s) {
  if (threadIdx.x == 0) { s[0] = INFINITY; s[1] = -INFINITY; }
}

// ---------------- global min/max of x/scales -------------------------------
__global__ void k_minmax(const float* __restrict__ x,
                         const float* __restrict__ scales,
                         float* __restrict__ s) {
  float lmn = INFINITY, lmx = -INFINITY;
  const long total = (long)Mtot * IN;
  const long stride = (long)gridDim.x * blockDim.x;
  for (long i = (long)blockIdx.x * blockDim.x + threadIdx.x; i < total; i += stride) {
    int k = (int)(i % IN);
    float v = x[i] / scales[k];
    lmn = fminf(lmn, v);
    lmx = fmaxf(lmx, v);
  }
  __shared__ float smn[256], smx[256];
  int tid = threadIdx.x;
  smn[tid] = lmn; smx[tid] = lmx;
  __syncthreads();
  for (int o = 128; o > 0; o >>= 1) {
    if (tid < o) {
      smn[tid] = fminf(smn[tid], smn[tid + o]);
      smx[tid] = fmaxf(smx[tid], smx[tid + o]);
    }
    __syncthreads();
  }
  if (tid == 0) { atomicMinF(&s[0], smn[0]); atomicMaxF(&s[1], smx[0]); }
}

// ---------------- derive scale / zero-point --------------------------------
__global__ void k_scale(float* s) {
  if (threadIdx.x == 0) {
    float mn = s[0], mx = s[1];
    float sc = (mx - mn) / QMAXF;
    s[2] = sc;
    s[3] = rintf(-mn / sc);   // RNE, matches jnp.round
  }
}

// ---------------- quantize: A[m, k<IN] = clip(round(xs/sc)+zp)-zp ----------
__global__ void k_quant(const float* __restrict__ x,
                        const float* __restrict__ scales,
                        const float* __restrict__ s,
                        _Float16* __restrict__ A) {
  long i = (long)blockIdx.x * blockDim.x + threadIdx.x;
  const long total = (long)Mtot * IN;
  if (i >= total) return;
  int  k = (int)(i % IN);
  long m = i / IN;
  float sc = s[2], zp = s[3];
  float xs = x[i] / scales[k];
  float q  = rintf(xs / sc) + zp;
  q = fminf(fmaxf(q, 0.0f), QMAXF);
  A[m * KEXT + k] = (_Float16)(q - zp);   // exact integer, |.| <= 255
}

// ---------------- weights: W_ext = [scale*quant_w | left_w] (f16) ----------
__global__ void k_wext(const float* __restrict__ qw,
                       const float* __restrict__ lw,
                       const float* __restrict__ s,
                       _Float16* __restrict__ W) {
  int i = blockIdx.x * blockDim.x + threadIdx.x;
  if (i >= OUT * KEXT) return;
  int k = i % KEXT, n = i / KEXT;
  float v = (k < IN) ? qw[n * IN + k] * s[2] : lw[n * RNK + (k - IN)];
  W[i] = (_Float16)v;
}

// ---------------- Rw = scale * right_w (f16) -------------------------------
__global__ void k_rw(const float* __restrict__ rw,
                     const float* __restrict__ s,
                     _Float16* __restrict__ Rw) {
  int i = blockIdx.x * blockDim.x + threadIdx.x;
  if (i >= RNK * IN) return;
  Rw[i] = (_Float16)(rw[i] * s[2]);
}

// ---------------- WMMA helpers ---------------------------------------------
__device__ inline v8f wmma16(v16h a, v16h b, v8f c) {
  return __builtin_amdgcn_wmma_f32_16x16x32_f16(
      /*neg_a=*/false, a, /*neg_b=*/false, b,
      /*c_mod=*/(short)0, c, /*reuse_a=*/false, /*reuse_b=*/false);
}

// Fragment layout (ISA 7.12.2, 16-bit A 16x32):
//   lane 0-15 : row = row0+lane,    K elements {0..7, 16..23}
//   lane 16-31: row = row0+lane-16, K elements {8..15, 24..31}
// -> per lane: 2 x 16-byte loads at row*stride + (lane>=16)*16B + {0,32}B.

// Load fragment straight from global (row-major, ldk halfs per row).
__device__ inline v16h load_frag(const _Float16* __restrict__ base,
                                 int row0, int ldk, int kb, int lane) {
  const _Float16* p = base + (long)(row0 + (lane & 15)) * ldk
                           + kb + ((lane >> 4) << 3);
  union { v16h v; unsigned u[8]; } r;
#pragma unroll
  for (int q = 0; q < 8; ++q) {
    int koff = ((q & 3) << 1) + ((q >> 2) << 4);   // 0,2,4,6,16,18,20,22
    r.u[q] = *(const unsigned*)(p + koff);
  }
  return r.v;
}

// Load fragment from an LDS-staged 128xROWH tile (K window already staged).
__device__ inline v16h load_frag_lds(const _Float16* buf, int row0, int lane) {
  const _Float16* p = buf + (row0 + (lane & 15)) * ROWH + ((lane >> 4) << 3);
  union { v16h v; uint4 q[2]; } r;
  r.q[0] = *(const uint4*)(p);
  r.q[1] = *(const uint4*)(p + 16);
  return r.v;
}

// Async-DMA one 128x32-half tile (global, row stride KEXT) into an LDS
// buffer (row stride ROWH). 512 16-byte chunks; 2 per thread (256 threads).
__device__ inline void issue_tile(const _Float16* __restrict__ gbase,
                                  int kb, _Float16* lbuf, int t) {
#pragma unroll
  for (int c = 0; c < 2; ++c) {
    int ch  = t + c * 256;                 // 0..511
    int row = ch >> 2;                     // 0..127
    int c16 = ch & 3;                      // 16B chunk within 64B row
    unsigned long long ga =
        (unsigned long long)(gbase + (long)row * KEXT + kb + c16 * 8);
    unsigned la = (unsigned)(unsigned long long)(lbuf + row * ROWH + c16 * 8);
    asm volatile("global_load_async_to_lds_b128 %0, %1, off"
                 :: "v"(la), "v"(ga) : "memory");
  }
}
__device__ inline void wait_async0() {
  asm volatile("s_wait_asynccnt 0x0" ::: "memory");
}

// ---------------- low-rank: A[:,1152:1216] = f16( A[:,0:1152] @ Rw^T ) -----
__global__ __launch_bounds__(256, 1)
void k_lowrank(const _Float16* __restrict__ A,
               const _Float16* __restrict__ Rw,
               _Float16* __restrict__ Aout) {
  int lane = threadIdx.x & 31;
  int wave = threadIdx.x >> 5;
  int m0 = blockIdx.x * 256 + wave * 32;   // each wave: 32 rows x 64 cols
  v8f acc[2][4] = {};
  for (int kb = 0; kb < IN; kb += 32) {
    v16h a0 = load_frag(A, m0,      KEXT, kb, lane);
    v16h a1 = load_frag(A, m0 + 16, KEXT, kb, lane);
    v16h b0 = load_frag(Rw,  0, IN, kb, lane);
    v16h b1 = load_frag(Rw, 16, IN, kb, lane);
    v16h b2 = load_frag(Rw, 32, IN, kb, lane);
    v16h b3 = load_frag(Rw, 48, IN, kb, lane);
    acc[0][0] = wmma16(a0, b0, acc[0][0]);
    acc[0][1] = wmma16(a0, b1, acc[0][1]);
    acc[0][2] = wmma16(a0, b2, acc[0][2]);
    acc[0][3] = wmma16(a0, b3, acc[0][3]);
    acc[1][0] = wmma16(a1, b0, acc[1][0]);
    acc[1][1] = wmma16(a1, b1, acc[1][1]);
    acc[1][2] = wmma16(a1, b2, acc[1][2]);
    acc[1][3] = wmma16(a1, b3, acc[1][3]);
  }
#pragma unroll
  for (int i = 0; i < 2; ++i)
#pragma unroll
    for (int j = 0; j < 4; ++j) {
      int n = j * 16 + (lane & 15);
#pragma unroll
      for (int r = 0; r < 8; ++r) {
        int m = m0 + i * 16 + r + ((lane >> 4) << 3);
        Aout[(long)m * KEXT + IN + n] = (_Float16)acc[i][j][r];
      }
    }
}

// ---------------- main GEMM: out = A_ext @ W_ext^T + bias ------------------
// Block: 256 thr = 8 waves (4 along M x 2 along N), tile 128x128.
// Double-buffered async LDS staging of A- and B-tiles; WMMA fed from LDS.
__global__ __launch_bounds__(256, 1)
void k_gemm(const _Float16* __restrict__ A,
            const _Float16* __restrict__ W,
            const float* __restrict__ bias,
            float* __restrict__ out) {
  __shared__ __align__(16) _Float16 lds[4 * TILEH];   // 40 KB
  _Float16* As[2] = { lds,             lds + TILEH     };
  _Float16* Bs[2] = { lds + 2 * TILEH, lds + 3 * TILEH };

  int t    = threadIdx.x;
  int lane = t & 31;
  int wave = t >> 5;
  int wm = wave & 3, wn = wave >> 2;
  int mblk = blockIdx.x * 128;
  int nblk = blockIdx.y * 128;
  const _Float16* Ag = A + (long)mblk * KEXT;
  const _Float16* Wg = W + (long)nblk * KEXT;

  // Prologue: stage tile 0.
  issue_tile(Ag, 0, As[0], t);
  issue_tile(Wg, 0, Bs[0], t);
  wait_async0();
  __syncthreads();

  v8f acc[2][4] = {};
  constexpr int NT = KEXT / 32;          // 38 K-steps
  for (int it = 0; it < NT; ++it) {
    int cur = it & 1, nxt = cur ^ 1;
    if (it + 1 < NT) {                   // overlap DMA of next tile with WMMA
      issue_tile(Ag, (it + 1) * 32, As[nxt], t);
      issue_tile(Wg, (it + 1) * 32, Bs[nxt], t);
    }
    v16h a0 = load_frag_lds(As[cur], wm * 32,      lane);
    v16h a1 = load_frag_lds(As[cur], wm * 32 + 16, lane);
    v16h b0 = load_frag_lds(Bs[cur], wn * 64,      lane);
    v16h b1 = load_frag_lds(Bs[cur], wn * 64 + 16, lane);
    v16h b2 = load_frag_lds(Bs[cur], wn * 64 + 32, lane);
    v16h b3 = load_frag_lds(Bs[cur], wn * 64 + 48, lane);
    acc[0][0] = wmma16(a0, b0, acc[0][0]);
    acc[0][1] = wmma16(a0, b1, acc[0][1]);
    acc[0][2] = wmma16(a0, b2, acc[0][2]);
    acc[0][3] = wmma16(a0, b3, acc[0][3]);
    acc[1][0] = wmma16(a1, b0, acc[1][0]);
    acc[1][1] = wmma16(a1, b1, acc[1][1]);
    acc[1][2] = wmma16(a1, b2, acc[1][2]);
    acc[1][3] = wmma16(a1, b3, acc[1][3]);
    if (it + 1 < NT) {
      wait_async0();      // next tile landed (this wave's chunks)
      __syncthreads();    // everyone's chunks landed; no one still reads 'nxt'
    }
  }

#pragma unroll
  for (int i = 0; i < 2; ++i)
#pragma unroll
    for (int j = 0; j < 4; ++j) {
      int n = nblk + wn * 64 + j * 16 + (lane & 15);
      float bv = bias[n];
#pragma unroll
      for (int r = 0; r < 8; ++r) {
        int m = mblk + wm * 32 + i * 16 + r + ((lane >> 4) << 3);
        out[(long)m * OUT + n] = acc[i][j][r] + bv;
      }
    }
}

// ---------------------------------------------------------------------------
extern "C" void kernel_launch(void* const* d_in, const int* in_sizes, int n_in,
                              void* d_out, int out_size, void* d_ws, size_t ws_size,
                              hipStream_t stream) {
  const float* x      = (const float*)d_in[0];
  const float* qw     = (const float*)d_in[1];
  const float* rw     = (const float*)d_in[2];
  const float* lw     = (const float*)d_in[3];
  const float* bias   = (const float*)d_in[4];
  const float* scales = (const float*)d_in[5];
  float* out = (float*)d_out;

  char* ws = (char*)d_ws;
  float*    s  = (float*)ws;                                        // 4 scalars
  _Float16* A  = (_Float16*)(ws + 256);                             // M x KEXT
  _Float16* W  = (_Float16*)(ws + 256 + (size_t)Mtot * KEXT * 2);   // OUT x KEXT
  _Float16* Rw = (_Float16*)(ws + 256 + (size_t)Mtot * KEXT * 2
                                       + (size_t)OUT  * KEXT * 2);  // RNK x IN

  k_init<<<1, 1, 0, stream>>>(s);
  k_minmax<<<2048, 256, 0, stream>>>(x, scales, s);
  k_scale<<<1, 1, 0, stream>>>(s);

  long total = (long)Mtot * IN;
  k_quant<<<(unsigned)((total + 255) / 256), 256, 0, stream>>>(x, scales, s, A);
  k_wext<<<(OUT * KEXT + 255) / 256, 256, 0, stream>>>(qw, lw, s, W);
  k_rw<<<(RNK * IN + 255) / 256, 256, 0, stream>>>(rw, s, Rw);

  k_lowrank<<<Mtot / 256, 256, 0, stream>>>(A, Rw, A);
  k_gemm<<<dim3(Mtot / 128, OUT / 128), 256, 0, stream>>>(A, W, bias, out);
}
